// Model_65171833749593
// MI455X (gfx1250) — matrix operand order repaired and checked by reading, not compiled, not run on hardware
//
#include <hip/hip_runtime.h>
#include <hip/hip_bf16.h>

// ---------------------------------------------------------------------------
// Types for CDNA5 WMMA (wave32, v_wmma_f32_16x16x32_bf16)
// ---------------------------------------------------------------------------
typedef __bf16 bf16_t;
typedef __attribute__((ext_vector_type(16))) __bf16 v16bf;
typedef __attribute__((ext_vector_type(8)))  __bf16 v8bf;
typedef __attribute__((ext_vector_type(8)))  float  v8f;

#define NP_N 100000
#define ND_N 50000
#define DIN_ 128
#define H_   128
#define E_   600000
#define EL_  500000

static __device__ __forceinline__ bf16_t f2bf(float f) {
  union { float f; unsigned u; } v; v.f = f;
  unsigned r = v.u + 0x7fffu + ((v.u >> 16) & 1u);   // round-to-nearest-even
  union { unsigned short s; bf16_t b; } o; o.s = (unsigned short)(r >> 16);
  return o.b;
}

static __device__ __forceinline__ v8f wmma_bf16(v16bf a, v16bf b, v8f c) {
  return __builtin_amdgcn_wmma_f32_16x16x32_bf16(
      /*neg_a=*/false, a, /*neg_b=*/false, b,
      /*c_mod=*/(short)0, c, /*reuse_a=*/false, /*reuse_b=*/false);
}

// A-fragment (bf16 16x32): per-lane chunks at +0 (K lo..lo+7) and +16 (K lo+16..lo+23)
static __device__ __forceinline__ v16bf ldA(const bf16_t* p) {
  v8bf a = *(const v8bf*)(p);
  v8bf b = *(const v8bf*)(p + 16);
  v16bf r;
#pragma unroll
  for (int i = 0; i < 8; ++i) { r[i] = a[i]; r[i + 8] = b[i]; }
  return r;
}

// B-fragment (bf16 32x16) from K-contiguous (transposed) weights: chunks at +0 (K lo..lo+7) and +16 (K lo+16..lo+23)
static __device__ __forceinline__ v16bf ldB(const bf16_t* p) {
  v8bf a = *(const v8bf*)(p);
  v8bf b = *(const v8bf*)(p + 16);
  v16bf r;
#pragma unroll
  for (int i = 0; i < 8; ++i) { r[i] = a[i]; r[i + 8] = b[i]; }
  return r;
}

// ---------------------------------------------------------------------------
// Graph kernels
// ---------------------------------------------------------------------------
__global__ void k_degree(const int* __restrict__ dst, float* __restrict__ deg, int E) {
  int t = blockIdx.x * blockDim.x + threadIdx.x;
  if (t < E) atomicAdd(&deg[dst[t]], 1.0f);
}

// one wave per edge; lane handles 4 contiguous floats (128/32)
__global__ void k_scatter(const float* __restrict__ xsrc, const int* __restrict__ src,
                          const int* __restrict__ dst, float* __restrict__ agg, int E) {
  int tid  = blockIdx.x * blockDim.x + threadIdx.x;
  int gw   = tid >> 5;
  int lane = tid & 31;
  if (gw >= E) return;
  int s = src[gw], d = dst[gw];
  float4 v = ((const float4*)(xsrc + (size_t)s * 128))[lane];
  float* o = agg + (size_t)d * 128 + lane * 4;
  atomicAdd(o + 0, v.x);
  atomicAdd(o + 1, v.y);
  atomicAdd(o + 2, v.z);
  atomicAdd(o + 3, v.w);
}

// fp32 [n,128] -> bf16 [n,128], optionally dividing each row by max(deg,1) (mean aggr)
__global__ void k_cvt_rows(const float* __restrict__ in, const float* __restrict__ deg,
                           bf16_t* __restrict__ out, int n) {
  int i = blockIdx.x * blockDim.x + threadIdx.x;
  if (i >= n) return;
  float s = 1.0f;
  if (deg) s = 1.0f / fmaxf(deg[i >> 7], 1.0f);
  out[i] = f2bf(in[i] * s);
}

// W [K,N] fp32 row-major -> WT [N,K] bf16 (K contiguous per output column)
__global__ void k_cvt_wt(const float* __restrict__ W, bf16_t* __restrict__ WT, int K, int N) {
  int i = blockIdx.x * blockDim.x + threadIdx.x;
  if (i >= K * N) return;
  int k = i / N, n = i % N;
  WT[(size_t)n * K + k] = f2bf(W[i]);
}

// ---------------------------------------------------------------------------
// Fused SAGE GEMM: out = relu( A @ Wl + b + X @ Wr ), A/X [n,128] bf16.
// Both weight matrices (2 x 32KB, K-contiguous) staged once per block into LDS;
// B fragments then come from ds_load_b128.  One wave -> 16 rows x 128 cols.
// ---------------------------------------------------------------------------
__global__ void k_sage_gemm(const bf16_t* __restrict__ A, const bf16_t* __restrict__ X,
                            const bf16_t* __restrict__ WlT, const bf16_t* __restrict__ WrT,
                            const float* __restrict__ bias,
                            float* __restrict__ outf, bf16_t* __restrict__ outb, int nrows) {
  __shared__ __align__(16) unsigned char smem_raw[2 * 128 * 128 * 2];  // 64KB
  bf16_t* sWl = (bf16_t*)smem_raw;
  bf16_t* sWr = sWl + 128 * 128;

  int tid  = blockIdx.x * blockDim.x + threadIdx.x;
  int gw   = tid >> 5;
  int row0 = gw * 16;
  bool active = (row0 < nrows);
  int lane = threadIdx.x & 31;
  int mloc = lane & 15;
  int hi   = lane >> 4;

  const bf16_t* arow = nullptr;
  const bf16_t* xrow = nullptr;
  if (active) {
    arow = A + (size_t)(row0 + mloc) * 128 + hi * 8;
    xrow = X + (size_t)(row0 + mloc) * 128 + hi * 8;
    // warm L2/WGP$ for this wave's 16 A/X rows while the block stages weights
    __builtin_prefetch(A + (size_t)(row0 + mloc) * 128 + hi * 64, 0, 3);
    __builtin_prefetch(X + (size_t)(row0 + mloc) * 128 + hi * 64, 0, 3);
  }

  // cooperative stage: 2 x 32KB = 4096 uint4
  {
    uint4* sv = (uint4*)smem_raw;
    const uint4* gl = (const uint4*)WlT;
    const uint4* gr = (const uint4*)WrT;
#pragma unroll
    for (int it = 0; it < 8; ++it) {
      int i = it * 256 + threadIdx.x;
      sv[i] = gl[i];
      sv[2048 + i] = gr[i];
    }
  }
  __syncthreads();
  if (!active) return;

  v8f acc[8];
#pragma unroll
  for (int t = 0; t < 8; ++t) acc[t] = (v8f)(0.0f);

#pragma unroll
  for (int ks = 0; ks < 4; ++ks) {
    int k0 = ks * 32;
    v16bf af = ldA(arow + k0);
    v16bf xf = ldA(xrow + k0);
#pragma unroll
    for (int t = 0; t < 8; ++t) {
      v16bf bl = ldB(sWl + (t * 16 + mloc) * 128 + k0 + hi * 8);
      v16bf br = ldB(sWr + (t * 16 + mloc) * 128 + k0 + hi * 8);
      acc[t] = wmma_bf16(af, bl, acc[t]);
      acc[t] = wmma_bf16(xf, br, acc[t]);
    }
  }

#pragma unroll
  for (int t = 0; t < 8; ++t) {
    int n = t * 16 + mloc;
    float bv = bias[n];
#pragma unroll
    for (int r = 0; r < 8; ++r) {
      float y = fmaxf(acc[t][r] + bv, 0.0f);
      size_t idx = (size_t)(row0 + hi * 8 + r) * 128 + n;
      if (outf) outf[idx] = y;
      if (outb) outb[idx] = f2bf(y);
    }
  }
}

// ---------------------------------------------------------------------------
// Fused decoder: per 16-edge tile:
//   y = relu(concat(zp[row], zd[col]) @ W1 + b1);  out = relu(y . w2 + b2)
// W1T (64KB) staged in LDS; random z-row gathers prefetched behind the staging.
// K=256 (8 bf16 WMMA steps), then in-register relu/dot + wave32 xor-reduction.
// ---------------------------------------------------------------------------
__global__ void k_decoder(const bf16_t* __restrict__ zp, const bf16_t* __restrict__ zd,
                          const int* __restrict__ row, const int* __restrict__ col,
                          const bf16_t* __restrict__ W1T, const float* __restrict__ b1,
                          const float* __restrict__ w2, const float* __restrict__ b2p,
                          float* __restrict__ out, int ntiles) {
  __shared__ __align__(16) unsigned char smem_raw[256 * 128 * 2];  // 64KB
  bf16_t* sW1 = (bf16_t*)smem_raw;

  int tid  = blockIdx.x * blockDim.x + threadIdx.x;
  int gw   = tid >> 5;
  bool active = (gw < ntiles);
  int lane = threadIdx.x & 31;
  int mloc = lane & 15;
  int hi   = lane >> 4;

  const bf16_t* prow = nullptr;
  const bf16_t* drow = nullptr;
  if (active) {
    int e = gw * 16 + mloc;
    prow = zp + (size_t)row[e] * 128;
    drow = zd + (size_t)col[e] * 128;
    // prefetch the 16 random gather rows (2 lines each) behind the LDS staging
    __builtin_prefetch(prow + hi * 64, 0, 3);
    __builtin_prefetch(drow + hi * 64, 0, 3);
  }

  // cooperative stage of W1T: 64KB = 4096 uint4
  {
    uint4* sv = (uint4*)smem_raw;
    const uint4* g = (const uint4*)W1T;
#pragma unroll
    for (int it = 0; it < 16; ++it) {
      int i = it * 256 + threadIdx.x;
      sv[i] = g[i];
    }
  }
  __syncthreads();
  if (!active) return;

  v8f acc[8];
#pragma unroll
  for (int t = 0; t < 8; ++t) acc[t] = (v8f)(0.0f);

#pragma unroll
  for (int ks = 0; ks < 8; ++ks) {
    int k0 = ks * 32;
    const bf16_t* srcp = (k0 < 128) ? prow : drow;
    int kk = (k0 < 128) ? k0 : (k0 - 128);
    v16bf af = ldA(srcp + kk + hi * 8);
#pragma unroll
    for (int t = 0; t < 8; ++t) {
      v16bf bf_ = ldB(sW1 + (t * 16 + mloc) * 256 + k0 + hi * 8);
      acc[t] = wmma_bf16(af, bf_, acc[t]);
    }
  }

  float p[8];
#pragma unroll
  for (int r = 0; r < 8; ++r) p[r] = 0.0f;
#pragma unroll
  for (int t = 0; t < 8; ++t) {
    int n = t * 16 + mloc;
    float b1v = b1[n], w2v = w2[n];
#pragma unroll
    for (int r = 0; r < 8; ++r)
      p[r] += fmaxf(acc[t][r] + b1v, 0.0f) * w2v;
  }
  // reduce over the 16 lanes of each half-wave (xor by 1,2,4,8 stays in-half)
#pragma unroll
  for (int r = 0; r < 8; ++r) {
#pragma unroll
    for (int off = 1; off < 16; off <<= 1)
      p[r] += __shfl_xor(p[r], off, 32);
  }
  if (mloc == 0) {
    float b2 = b2p[0];
#pragma unroll
    for (int r = 0; r < 8; ++r)
      out[gw * 16 + hi * 8 + r] = fmaxf(p[r] + b2, 0.0f);
  }
}

// ---------------------------------------------------------------------------
// Host orchestration
// ---------------------------------------------------------------------------
static inline size_t alloc_off(size_t& cur, size_t bytes) {
  size_t o = cur;
  cur = (cur + bytes + 255) & ~(size_t)255;
  return o;
}

extern "C" void kernel_launch(void* const* d_in, const int* in_sizes, int n_in,
                              void* d_out, int out_size, void* d_ws, size_t ws_size,
                              hipStream_t stream) {
  const float* x_p    = (const float*)d_in[0];
  const float* x_d    = (const float*)d_in[1];
  const int*   src_pd = (const int*)d_in[2];
  const int*   dst_pd = (const int*)d_in[3];
  const int*   src_dp = (const int*)d_in[4];
  const int*   dst_dp = (const int*)d_in[5];
  const int*   row    = (const int*)d_in[6];
  const int*   col    = (const int*)d_in[7];
  const float* Wl1_pd = (const float*)d_in[8];
  const float* bl1_pd = (const float*)d_in[9];
  const float* Wr1_pd = (const float*)d_in[10];
  const float* Wl1_dp = (const float*)d_in[11];
  const float* bl1_dp = (const float*)d_in[12];
  const float* Wr1_dp = (const float*)d_in[13];
  const float* Wl2_pd = (const float*)d_in[14];
  const float* bl2_pd = (const float*)d_in[15];
  const float* Wr2_pd = (const float*)d_in[16];
  const float* Wl2_dp = (const float*)d_in[17];
  const float* bl2_dp = (const float*)d_in[18];
  const float* Wr2_dp = (const float*)d_in[19];
  const float* dec_W1 = (const float*)d_in[20];
  const float* dec_b1 = (const float*)d_in[21];
  const float* dec_W2 = (const float*)d_in[22];
  const float* dec_b2 = (const float*)d_in[23];
  float* outp = (float*)d_out;

  char* ws = (char*)d_ws;
  size_t cur = 0;
  float*  agg_p = (float*)(ws + alloc_off(cur, (size_t)NP_N * 128 * 4));
  float*  agg_d = (float*)(ws + alloc_off(cur, (size_t)ND_N * 128 * 4));
  float*  deg_p = (float*)(ws + alloc_off(cur, (size_t)NP_N * 4));
  float*  deg_d = (float*)(ws + alloc_off(cur, (size_t)ND_N * 4));
  float*  h_p   = (float*)(ws + alloc_off(cur, (size_t)NP_N * 128 * 4));
  float*  h_d   = (float*)(ws + alloc_off(cur, (size_t)ND_N * 128 * 4));
  bf16_t* Abf_p = (bf16_t*)(ws + alloc_off(cur, (size_t)NP_N * 128 * 2));
  bf16_t* Abf_d = (bf16_t*)(ws + alloc_off(cur, (size_t)ND_N * 128 * 2));
  bf16_t* Xbf_p = (bf16_t*)(ws + alloc_off(cur, (size_t)NP_N * 128 * 2));
  bf16_t* Xbf_d = (bf16_t*)(ws + alloc_off(cur, (size_t)ND_N * 128 * 2));
  bf16_t* Hbf_p = (bf16_t*)(ws + alloc_off(cur, (size_t)NP_N * 128 * 2));
  bf16_t* Hbf_d = (bf16_t*)(ws + alloc_off(cur, (size_t)ND_N * 128 * 2));
  bf16_t* Zbf_p = (bf16_t*)(ws + alloc_off(cur, (size_t)NP_N * 128 * 2));
  bf16_t* Zbf_d = (bf16_t*)(ws + alloc_off(cur, (size_t)ND_N * 128 * 2));
  bf16_t* wt[8];
  for (int i = 0; i < 8; ++i) wt[i] = (bf16_t*)(ws + alloc_off(cur, (size_t)128 * 128 * 2));
  bf16_t* W1T = (bf16_t*)(ws + alloc_off(cur, (size_t)256 * 128 * 2));

  const int TB = 256;
  // -------- zero accumulators / degrees --------
  hipMemsetAsync(agg_p, 0, (size_t)NP_N * 128 * 4, stream);
  hipMemsetAsync(agg_d, 0, (size_t)ND_N * 128 * 4, stream);
  hipMemsetAsync(deg_p, 0, (size_t)NP_N * 4, stream);
  hipMemsetAsync(deg_d, 0, (size_t)ND_N * 4, stream);

  // -------- degrees (same edge lists both layers) --------
  k_degree<<<(E_ + TB - 1) / TB, TB, 0, stream>>>(dst_pd, deg_d, E_);
  k_degree<<<(E_ + TB - 1) / TB, TB, 0, stream>>>(dst_dp, deg_p, E_);

  // -------- weight transpose+convert (bf16, K-contiguous) --------
  int wtb = (128 * 128 + TB - 1) / TB;
  k_cvt_wt<<<wtb, TB, 0, stream>>>(Wl1_pd, wt[0], 128, 128);
  k_cvt_wt<<<wtb, TB, 0, stream>>>(Wr1_pd, wt[1], 128, 128);
  k_cvt_wt<<<wtb, TB, 0, stream>>>(Wl1_dp, wt[2], 128, 128);
  k_cvt_wt<<<wtb, TB, 0, stream>>>(Wr1_dp, wt[3], 128, 128);
  k_cvt_wt<<<wtb, TB, 0, stream>>>(Wl2_pd, wt[4], 128, 128);
  k_cvt_wt<<<wtb, TB, 0, stream>>>(Wr2_pd, wt[5], 128, 128);
  k_cvt_wt<<<wtb, TB, 0, stream>>>(Wl2_dp, wt[6], 128, 128);
  k_cvt_wt<<<wtb, TB, 0, stream>>>(Wr2_dp, wt[7], 128, 128);
  k_cvt_wt<<<(256 * 128 + TB - 1) / TB, TB, 0, stream>>>(dec_W1, W1T, 256, 128);

  // -------- layer 1: scatter-sum then mean+convert --------
  int sc_blocks = (E_ * 32 + TB - 1) / TB;
  k_scatter<<<sc_blocks, TB, 0, stream>>>(x_p, src_pd, dst_pd, agg_d, E_);
  k_scatter<<<sc_blocks, TB, 0, stream>>>(x_d, src_dp, dst_dp, agg_p, E_);

  int np_el = NP_N * 128, nd_el = ND_N * 128;
  k_cvt_rows<<<(np_el + TB - 1) / TB, TB, 0, stream>>>(x_p,   nullptr, Xbf_p, np_el);
  k_cvt_rows<<<(nd_el + TB - 1) / TB, TB, 0, stream>>>(x_d,   nullptr, Xbf_d, nd_el);
  k_cvt_rows<<<(np_el + TB - 1) / TB, TB, 0, stream>>>(agg_p, deg_p,   Abf_p, np_el);
  k_cvt_rows<<<(nd_el + TB - 1) / TB, TB, 0, stream>>>(agg_d, deg_d,   Abf_d, nd_el);

  // -------- layer 1 WMMA GEMMs (dual fp32 + bf16 output) --------
  int gp_blocks = ((NP_N / 16) * 32 + TB - 1) / TB;   // 6250 waves
  int gd_blocks = ((ND_N / 16) * 32 + TB - 1) / TB;   // 3125 waves
  k_sage_gemm<<<gd_blocks, TB, 0, stream>>>(Abf_d, Xbf_d, wt[0], wt[1], bl1_pd, h_d, Hbf_d, ND_N);
  k_sage_gemm<<<gp_blocks, TB, 0, stream>>>(Abf_p, Xbf_p, wt[2], wt[3], bl1_dp, h_p, Hbf_p, NP_N);

  // -------- layer 2: re-zero, scatter h, mean+convert --------
  hipMemsetAsync(agg_p, 0, (size_t)NP_N * 128 * 4, stream);
  hipMemsetAsync(agg_d, 0, (size_t)ND_N * 128 * 4, stream);
  k_scatter<<<sc_blocks, TB, 0, stream>>>(h_p, src_pd, dst_pd, agg_d, E_);
  k_scatter<<<sc_blocks, TB, 0, stream>>>(h_d, src_dp, dst_dp, agg_p, E_);
  k_cvt_rows<<<(np_el + TB - 1) / TB, TB, 0, stream>>>(agg_p, deg_p, Abf_p, np_el);
  k_cvt_rows<<<(nd_el + TB - 1) / TB, TB, 0, stream>>>(agg_d, deg_d, Abf_d, nd_el);

  // -------- layer 2 WMMA GEMMs (bf16-only output for decoder gathers) --------
  k_sage_gemm<<<gd_blocks, TB, 0, stream>>>(Abf_d, Hbf_d, wt[4], wt[5], bl2_pd, nullptr, Zbf_d, ND_N);
  k_sage_gemm<<<gp_blocks, TB, 0, stream>>>(Abf_p, Hbf_p, wt[6], wt[7], bl2_dp, nullptr, Zbf_p, NP_N);

  // -------- fused decoder --------
  int ntiles = EL_ / 16;                               // 31250 waves
  int dec_blocks = (ntiles * 32 + TB - 1) / TB;
  k_decoder<<<dec_blocks, TB, 0, stream>>>(Zbf_p, Zbf_d, row, col, W1T,
                                           dec_b1, dec_W2, dec_b2, outp, ntiles);
}
